// Encoder_73658689126885
// MI455X (gfx1250) — matrix operand order repaired and checked by reading, not compiled, not run on hardware
//
#include <hip/hip_runtime.h>
#include <hip/hip_bf16.h>
#include <math.h>

typedef __attribute__((ext_vector_type(16))) _Float16 v16h;
typedef __attribute__((ext_vector_type(8)))  _Float16 v8h;
typedef __attribute__((ext_vector_type(8)))  float    v8f;
typedef __attribute__((ext_vector_type(4)))  float    f4;
typedef __attribute__((ext_vector_type(4)))  int      v4i;

#define BN_EPS 1e-5f
#define NB 16
#define NT 2000

#if __has_builtin(__builtin_amdgcn_global_load_async_to_lds_b128)
#define HAVE_ASYNC_COPY 1
#else
#define HAVE_ASYNC_COPY 0
#endif

__device__ __forceinline__ void wait_async0() {
#if __has_builtin(__builtin_amdgcn_s_wait_asynccnt)
  __builtin_amdgcn_s_wait_asynccnt(0);
#else
  asm volatile("s_wait_asynccnt 0x0" ::: "memory");
#endif
}

#if HAVE_ASYNC_COPY
// 16-byte async DMA global->LDS. Param 0 is a typed int4 pointer in the
// global address space (probe-confirmed by the compiler diagnostic); the
// LDS destination must be a real addrspace(3) pointer (addrspacecast).
__device__ __forceinline__ void async_copy_b128(const float* gsrc, float* ldst) {
  typedef __attribute__((address_space(1))) v4i* gp_t;
  typedef __attribute__((address_space(3))) v4i* lp_t;
  __builtin_amdgcn_global_load_async_to_lds_b128(
      (gp_t)(unsigned long long)gsrc, (lp_t)ldst, 0, 0);
}
#endif

// ---------------------------------------------------------------- utilities

__global__ void zero_kernel(float* __restrict__ p, int n) {
  int i = blockIdx.x * blockDim.x + threadIdx.x;
  if (i < n) p[i] = 0.f;
}

// One block per (b,c) slab of perBC contiguous elements; accumulate
// sum / sumsq into stats[c] / stats[32+c] with float atomics.
__global__ void stats_kernel(const float* __restrict__ x, float* __restrict__ stats,
                             int C, int perBC) {
  int bc = blockIdx.x;
  int c  = bc % C;
  const float* p = x + (long)bc * perBC;
  float s = 0.f, s2 = 0.f;
  for (int i = threadIdx.x; i < perBC; i += 256) {
    float v = p[i];
    s += v; s2 += v * v;
  }
  __shared__ float sh[256], sh2[256];
  sh[threadIdx.x] = s; sh2[threadIdx.x] = s2;
  __syncthreads();
  for (int st = 128; st > 0; st >>= 1) {
    if (threadIdx.x < st) {
      sh[threadIdx.x]  += sh[threadIdx.x + st];
      sh2[threadIdx.x] += sh2[threadIdx.x + st];
    }
    __syncthreads();
  }
  if (threadIdx.x == 0) {
    atomicAdd(&stats[c],      sh[0]);
    atomicAdd(&stats[32 + c], sh2[0]);
  }
}

// In-place training-mode BN (+ PReLU) using per-channel sum/sumsq.
__global__ void bn_prelu_kernel(float* __restrict__ x, const float* __restrict__ stats,
                                const float* __restrict__ g, const float* __restrict__ bb,
                                const float* __restrict__ aPtr,
                                int C, int LT, long total, float invCnt) {
  long idx = (long)blockIdx.x * blockDim.x + threadIdx.x;
  if (idx >= total) return;
  int c = (int)((idx / LT) % C);
  float mean = stats[c] * invCnt;
  float var  = stats[32 + c] * invCnt - mean * mean;
  float inv  = rsqrtf(var + BN_EPS);
  float y = (x[idx] - mean) * inv * g[c] + bb[c];
  float a = aPtr[0];
  x[idx] = (y >= 0.f) ? y : a * y;
}

// ---------------------------------------------------------------- local conv
// out[b,c,l,t] = bias[c,l] + sum_{kc,i,j} xn[b,kc, l*strideH+i-ph, t+j-pw] *
//                W[((kc*kh+i)*kw+j)*Cout*L + c*L + l]
// xn = input optionally normalized on the fly (inStats != nullptr).
// Padding contributes 0 (applied AFTER normalization, matching the reference).
__global__ void local_conv_kernel(const float* __restrict__ xin, const float* __restrict__ W,
                                  const float* __restrict__ bias, float* __restrict__ out,
                                  const float* __restrict__ inStats,
                                  const float* __restrict__ inG, const float* __restrict__ inB,
                                  float inInvCnt,
                                  int Cin, int Hin, int Cout, int L,
                                  int kh, int kw, int strideH, int ph, int pw) {
  long total = (long)NB * Cout * L * NT;
  long idx = (long)blockIdx.x * blockDim.x + threadIdx.x;
  if (idx >= total) return;
  int t = (int)(idx % NT);
  long r = idx / NT;
  int l = (int)(r % L); r /= L;
  int c = (int)(r % Cout);
  int b = (int)(r / Cout);

  float acc = bias[c * L + l];
  for (int kc = 0; kc < Cin; ++kc) {
    float sc = 1.f, so = 0.f;
    if (inStats) {
      float mean = inStats[kc] * inInvCnt;
      float var  = inStats[32 + kc] * inInvCnt - mean * mean;
      float inv  = rsqrtf(var + BN_EPS);
      sc = inG[kc] * inv;
      so = inB[kc] - mean * sc;
    }
    const float* xb = xin + ((long)b * Cin + kc) * Hin * NT;
    for (int i = 0; i < kh; ++i) {
      int hh = l * strideH + i - ph;
      if ((unsigned)hh < (unsigned)Hin) {
        for (int j = 0; j < kw; ++j) {
          int tt = t + j - pw;
          if ((unsigned)tt < (unsigned)NT) {
            float xv = xb[(long)hh * NT + tt] * sc + so;
            acc += xv * W[(((kc * kh + i) * kw + j) * Cout + c) * L + l];
          }
        }
      }
    }
  }
  out[idx] = acc;
}

// ---------------------------------------------------------------- lin_in
// x_in[b,t, o*2+g] = PReLU( b_in[g,o] + sum_i cat[b,t, g*90+i] * Win[g,i,o] )
// cat g=0: erb_e3[b,c,l,t], i=c*6+l  ; g=1: com_e3[b,c,l,t], i=c*10+l
__global__ void lin_in_kernel(const float* __restrict__ erb3, const float* __restrict__ com3,
                              const float* __restrict__ Win, const float* __restrict__ bin,
                              const float* __restrict__ aPtr, float* __restrict__ xin) {
  long idx = (long)blockIdx.x * blockDim.x + threadIdx.x;
  if (idx >= (long)NB * NT * 48) return;
  int n = (int)(idx % 48);
  long bt = idx / 48;
  int t = (int)(bt % NT);
  int b = (int)(bt / NT);
  int g = n & 1, o = n >> 1;
  float acc = bin[g * 24 + o];
  const float* Wg = Win + (long)(g * 90) * 24 + o;  // stride 24 over i
  if (g == 0) {
    for (int c = 0; c < 15; ++c)
      for (int l = 0; l < 6; ++l)
        acc += erb3[(((long)b * 15 + c) * 6 + l) * NT + t] * Wg[(c * 6 + l) * 24];
  } else {
    for (int c = 0; c < 9; ++c)
      for (int l = 0; l < 10; ++l)
        acc += com3[(((long)b * 9 + c) * 10 + l) * NT + t] * Wg[(c * 10 + l) * 24];
  }
  float a = aPtr[0];
  xin[bt * 48 + n] = (acc >= 0.f) ? acc : a * acc;
}

// ---------------------------------------------------------------- WMMA helpers
// A-fragment (16x32 f16): lane(L) elem(e) -> K = e + 8*(e>=8) + 8*(L>=16)
//   => per lane, two contiguous runs of 8: [8*hi, +8) and [16+8*hi, +8)
// B-fragment (32x16 f16): lane elem e -> K = e + 8*(e>=8) + 8*(L>=16) (same as A), N = L%16
//   => per lane, two contiguous runs of 8: [8*hi, +8) and [16+8*hi, +8)

// gi[m,n] = bih[n] + sum_k xin[m,k] * Wih[n,k]   (M=32000, N=144, K=48)
// One wave per 16x16 tile; K chunks [0,32) and [32,48) (rest zero-padded).
// All fragment loads are contiguous b128 loads; padding is branchless.
__global__ __launch_bounds__(32) void gi_wmma_kernel(const float* __restrict__ xin,
                                                     const float* __restrict__ Wih,
                                                     const float* __restrict__ bih,
                                                     float* __restrict__ gi) {
  const int m0 = blockIdx.x * 16;
  const int n0 = blockIdx.y * 16;
  const int lane = threadIdx.x;
  const int lm = lane & 15, hi = lane >> 4;

  const float* arow = xin + (long)(m0 + lm) * 48;
  const float* brow = Wih + (long)(n0 + lm) * 48;
  f4 aA = *(const f4*)(arow + hi * 8);
  f4 aB = *(const f4*)(arow + hi * 8 + 4);
  f4 aC = *(const f4*)(arow + 16 + hi * 8);
  f4 aD = *(const f4*)(arow + 16 + hi * 8 + 4);
  f4 aE = *(const f4*)(arow + 32 + hi * 8);
  f4 aF = *(const f4*)(arow + 32 + hi * 8 + 4);
  f4 bA = *(const f4*)(brow + hi * 8);
  f4 bB = *(const f4*)(brow + hi * 8 + 4);
  f4 bC = *(const f4*)(brow + 16 + hi * 8);
  f4 bD = *(const f4*)(brow + 16 + hi * 8 + 4);
  f4 bE = *(const f4*)(brow + 32 + hi * 8);
  f4 bF = *(const f4*)(brow + 32 + hi * 8 + 4);
  f4 bG = *(const f4*)(brow + 40);
  f4 bH = *(const f4*)(brow + 44);
  const float z = hi ? 0.f : 1.f;   // B chunk-1 pad (K>=48) for upper lane half

  v16h a0, a1, b0, b1;
#pragma unroll
  for (int e = 0; e < 4; ++e) {
    a0[e]      = (_Float16)aA[e];
    a0[4 + e]  = (_Float16)aB[e];
    a0[8 + e]  = (_Float16)aC[e];
    a0[12 + e] = (_Float16)aD[e];
    a1[e]      = (_Float16)aE[e];
    a1[4 + e]  = (_Float16)aF[e];
    a1[8 + e]  = (_Float16)0.f;       // K = 48..63 pad
    a1[12 + e] = (_Float16)0.f;
    b0[e]      = (_Float16)bA[e];
    b0[4 + e]  = (_Float16)bB[e];
    b0[8 + e]  = (_Float16)bC[e];
    b0[12 + e] = (_Float16)bD[e];
    b1[e]      = (_Float16)bE[e];
    b1[4 + e]  = (_Float16)bF[e];
    b1[8 + e]  = (_Float16)(z * bG[e]);
    b1[12 + e] = (_Float16)(z * bH[e]);
  }
  v8f acc = {};
  acc = __builtin_amdgcn_wmma_f32_16x16x32_f16(false, a0, false, b0,
                                               (short)0, acc, false, false);
  acc = __builtin_amdgcn_wmma_f32_16x16x32_f16(false, a1, false, b1,
                                               (short)0, acc, false, false);
#pragma unroll
  for (int r = 0; r < 8; ++r) {
    int m = m0 + r + 8 * hi;     // D: VGPR r -> M = r + 8*(lane>=16), N = lane%16
    int n = n0 + lm;
    gi[(long)m * 144 + n] = acc[r] + bih[n];
  }
}

// ---------------------------------------------------------------- GRU scan
// Single workgroup, 12 waves. Waves 0..8 own one 16-wide column tile of
// gh = h @ Whh^T (two v_wmma_f32_16x16x32_f16 / step); all 384 lanes do gates.
// h lives in LDS as f32 (gates) + zero-padded f16 (WMMA A fragments).
// gi[t+1] is staged into LDS with async global->LDS copies (double buffer)
// so the serial chain never waits on L2 for the gate inputs.
__global__ __launch_bounds__(384) void gru_kernel(const float* __restrict__ gi,
                                                  const float* __restrict__ Whh,
                                                  const float* __restrict__ bhh,
                                                  float* __restrict__ ys) {
  __shared__ float    hf[16 * 48];
  __shared__ _Float16 h16[16 * 64];   // K padded 48 -> 64 (zeros, set once)
  __shared__ float    gh[16 * 144];
#if HAVE_ASYNC_COPY
  __shared__ float    gibuf[2][16 * 144];
#endif
  const int tid = threadIdx.x;
  const int wave = tid >> 5;
  const int lane = tid & 31;
  const int lm = lane & 15, hi = lane >> 4;
  const int n0 = wave * 16;

  // Preload Whh^T B-fragments (two K=32 chunks) into registers, branchless.
  v16h b0 = {}, b1 = {};
  if (wave < 9) {
    const float* wrow = Whh + (long)(n0 + lm) * 48;
    f4 bA = *(const f4*)(wrow + hi * 8);
    f4 bB = *(const f4*)(wrow + hi * 8 + 4);
    f4 bC = *(const f4*)(wrow + 16 + hi * 8);
    f4 bD = *(const f4*)(wrow + 16 + hi * 8 + 4);
    f4 bE = *(const f4*)(wrow + 32 + hi * 8);
    f4 bF = *(const f4*)(wrow + 32 + hi * 8 + 4);
    f4 bG = *(const f4*)(wrow + 40);
    f4 bH = *(const f4*)(wrow + 44);
    const float z = hi ? 0.f : 1.f;
#pragma unroll
    for (int e = 0; e < 4; ++e) {
      b0[e]      = (_Float16)bA[e];
      b0[4 + e]  = (_Float16)bB[e];
      b0[8 + e]  = (_Float16)bC[e];
      b0[12 + e] = (_Float16)bD[e];
      b1[e]      = (_Float16)bE[e];
      b1[4 + e]  = (_Float16)bF[e];
      b1[8 + e]  = (_Float16)(z * bG[e]);
      b1[12 + e] = (_Float16)(z * bH[e]);
    }
  }
  for (int i = tid; i < 16 * 48; i += 384) hf[i] = 0.f;
  for (int i = tid; i < 16 * 64; i += 384) h16[i] = (_Float16)0.f;
#if HAVE_ASYNC_COPY
  for (int i = tid; i < 16 * 144; i += 384) {   // stage t=0 synchronously
    int b = i / 144, n = i - b * 144;
    gibuf[0][i] = gi[((long)b * NT) * 144 + n];
  }
#endif
  __syncthreads();

  for (int t = 0; t < NT; ++t) {
    const int cur = t & 1;
#if HAVE_ASYNC_COPY
    if (t + 1 < NT) {   // kick async DMA of gi[t+1] into the other buffer
      for (int q = tid; q < 576; q += 384) {        // 576 x float4
        int b = q / 36, c4 = q - b * 36;
        const float* gsrc = gi + ((long)b * NT + t + 1) * 144 + c4 * 4;
        float* ldst = &gibuf[cur ^ 1][b * 144 + c4 * 4];
        async_copy_b128(gsrc, ldst);
      }
    }
#endif
    if (wave < 9) {   // wave-uniform guard: EXEC all ones inside
      // A fragments: four contiguous 16B LDS reads (pad rows are zeros).
      v8h p0 = *(const v8h*)(h16 + lm * 64 + hi * 8);
      v8h p1 = *(const v8h*)(h16 + lm * 64 + 16 + hi * 8);
      v8h p2 = *(const v8h*)(h16 + lm * 64 + 32 + hi * 8);
      v8h p3 = *(const v8h*)(h16 + lm * 64 + 48 + hi * 8);   // zeros
      v16h a0 = __builtin_shufflevector(p0, p1, 0, 1, 2, 3, 4, 5, 6, 7,
                                        8, 9, 10, 11, 12, 13, 14, 15);
      v16h a1 = __builtin_shufflevector(p2, p3, 0, 1, 2, 3, 4, 5, 6, 7,
                                        8, 9, 10, 11, 12, 13, 14, 15);
      v8f acc = {};
      acc = __builtin_amdgcn_wmma_f32_16x16x32_f16(false, a0, false, b0,
                                                   (short)0, acc, false, false);
      acc = __builtin_amdgcn_wmma_f32_16x16x32_f16(false, a1, false, b1,
                                                   (short)0, acc, false, false);
#pragma unroll
      for (int r = 0; r < 8; ++r)
        gh[(r + 8 * hi) * 144 + n0 + lm] = acc[r];
    }
    __syncthreads();

    for (int idx = tid; idx < 768; idx += 384) {
      int b = idx / 48, n = idx - b * 48;
#if HAVE_ASYNC_COPY
      const float* gir = &gibuf[cur][b * 144];
#else
      const float* gir = gi + ((long)b * NT + t) * 144;
#endif
      float ir = gir[n], iz = gir[48 + n], inn = gir[96 + n];
      float hr = gh[b * 144 + n]      + bhh[n];
      float hz = gh[b * 144 + 48 + n] + bhh[48 + n];
      float hn = gh[b * 144 + 96 + n] + bhh[96 + n];
      float rg = 1.f / (1.f + expf(-(ir + hr)));
      float zg = 1.f / (1.f + expf(-(iz + hz)));
      float ng = tanhf(inn + rg * hn);
      float hp = hf[b * 48 + n];
      float hnew = (1.f - zg) * ng + zg * hp;
      hf[b * 48 + n] = hnew;
      h16[b * 64 + n] = (_Float16)hnew;
      ys[((long)b * NT + t) * 48 + n] = hnew;
#if !HAVE_ASYNC_COPY
      if (t + 1 < NT) __builtin_prefetch(gir + 144, 0, 1);
#endif
    }
#if HAVE_ASYNC_COPY
    wait_async0();
#endif
    __syncthreads();
  }
}

// ---------------------------------------------------------------- lin_out
// out[b,t, o*2+g] = PReLU( bout[g,o] + sum_i (ys+xin)[b,t, g*24+i] * Wout[g,i,o] )
__global__ void lin_out_kernel(const float* __restrict__ ys, const float* __restrict__ xin,
                               const float* __restrict__ Wout, const float* __restrict__ bout,
                               const float* __restrict__ aPtr, float* __restrict__ out) {
  long idx = (long)blockIdx.x * blockDim.x + threadIdx.x;
  if (idx >= (long)NB * NT * 80) return;
  int n = (int)(idx % 80);
  long bt = idx / 80;
  int g = n & 1, o = n >> 1;
  float acc = bout[g * 40 + o];
  for (int i = 0; i < 24; ++i) {
    float xv = ys[bt * 48 + g * 24 + i] + xin[bt * 48 + g * 24 + i];
    acc += xv * Wout[(g * 24 + i) * 40 + o];
  }
  float a = aPtr[0];
  out[bt * 80 + n] = (acc >= 0.f) ? acc : a * acc;
}

// ---------------------------------------------------------------- host glue

extern "C" void kernel_launch(void* const* d_in, const int* in_sizes, int n_in,
                              void* d_out, int out_size, void* d_ws, size_t ws_size,
                              hipStream_t stream) {
  (void)in_sizes; (void)n_in; (void)out_size; (void)ws_size;
  const float* erb_feat = (const float*)d_in[0];
  const float* com_feat = (const float*)d_in[1];
  const float* erb_g  = (const float*)d_in[2];
  const float* erb_b  = (const float*)d_in[3];
  const float* com_g  = (const float*)d_in[4];
  const float* com_b  = (const float*)d_in[5];
  const float* ce1_W  = (const float*)d_in[6];
  const float* ce1_b  = (const float*)d_in[7];
  const float* ce1_g  = (const float*)d_in[8];
  const float* ce1_bb = (const float*)d_in[9];
  const float* ce2_W  = (const float*)d_in[10];
  const float* ce2_b  = (const float*)d_in[11];
  const float* ce2_g  = (const float*)d_in[12];
  const float* ce2_bb = (const float*)d_in[13];
  const float* ce3_W  = (const float*)d_in[14];
  const float* ce3_b  = (const float*)d_in[15];
  const float* ce3_g  = (const float*)d_in[16];
  const float* ce3_bb = (const float*)d_in[17];
  const float* cc1_W  = (const float*)d_in[18];
  const float* cc1_b  = (const float*)d_in[19];
  const float* cc1_g  = (const float*)d_in[20];
  const float* cc1_bb = (const float*)d_in[21];
  const float* cc2_W  = (const float*)d_in[22];
  const float* cc2_b  = (const float*)d_in[23];
  const float* cc2_g  = (const float*)d_in[24];
  const float* cc2_bb = (const float*)d_in[25];
  const float* cc3_W  = (const float*)d_in[26];
  const float* cc3_b  = (const float*)d_in[27];
  const float* cc3_g  = (const float*)d_in[28];
  const float* cc3_bb = (const float*)d_in[29];
  const float* a_conv = (const float*)d_in[30];
  const float* lin_in_W  = (const float*)d_in[31];
  const float* lin_in_b  = (const float*)d_in[32];
  const float* a_in      = (const float*)d_in[33];
  const float* gru_Wih   = (const float*)d_in[34];
  const float* gru_Whh   = (const float*)d_in[35];
  const float* gru_bih   = (const float*)d_in[36];
  const float* gru_bhh   = (const float*)d_in[37];
  const float* lin_out_W = (const float*)d_in[38];
  const float* lin_out_b = (const float*)d_in[39];
  const float* a_out     = (const float*)d_in[40];

  float* out = (float*)d_out;
  float* CAT  = out;                  // [16,2000,80]
  float* ERB1 = out + 2560000;        // [16,8,36,2000]
  float* ERB2 = out + 11776000;       // [16,12,18,2000]
  float* ERB3 = out + 18688000;       // [16,15,6,2000]

  float* ws = (float*)d_ws;
  float* stats = ws;                  // 1024 floats: 8 slots of 64
  float* S_ERB = stats + 0 * 64;
  float* S_COM = stats + 1 * 64;
  float* S_CE1 = stats + 2 * 64;
  float* S_CE2 = stats + 3 * 64;
  float* S_CE3 = stats + 4 * 64;
  float* S_CC1 = stats + 5 * 64;
  float* S_CC2 = stats + 6 * 64;
  float* S_CC3 = stats + 7 * 64;
  float* C1  = ws + 1024;             // com_e1 raw/norm [16,4,80,2000]  (10.24M)
  float* C2  = C1 + 10240000;         // com_e2          [16,8,40,2000]  (10.24M)
  float* C3  = C1;                    // com_e3 reuses C1 (2.88M)
  float* XIN = C1 + 2880000;          // x_in  [32000,48]  (1.536M)
  float* GI  = C1 + 4416000;          // gi    [32000,144] (4.608M)
  float* YS  = C2;                    // ys reuses C2 (1.536M)

  // 0) reset stats accumulators (fresh every call -> graph-replay safe)
  zero_kernel<<<4, 256, 0, stream>>>(stats, 1024);

  // 1) input BN stats
  stats_kernel<<<16, 256, 0, stream>>>(erb_feat, S_ERB, 1, 36 * NT);
  stats_kernel<<<32, 256, 0, stream>>>(com_feat, S_COM, 2, 320 * NT);

  // 2) erb branch
  local_conv_kernel<<<36000, 256, 0, stream>>>(erb_feat, ce1_W, ce1_b, ERB1,
      S_ERB, erb_g, erb_b, 1.f / (16.f * 36.f * 2000.f),
      1, 36, 8, 36, 2, 2, 1, 1, 1);
  stats_kernel<<<128, 256, 0, stream>>>(ERB1, S_CE1, 8, 36 * NT);
  bn_prelu_kernel<<<36000, 256, 0, stream>>>(ERB1, S_CE1, ce1_g, ce1_bb, a_conv,
      8, 36 * NT, 9216000L, 1.f / 1152000.f);

  local_conv_kernel<<<27000, 256, 0, stream>>>(ERB1, ce2_W, ce2_b, ERB2,
      nullptr, nullptr, nullptr, 0.f,
      8, 36, 12, 18, 4, 1, 2, 2, 0);
  stats_kernel<<<192, 256, 0, stream>>>(ERB2, S_CE2, 12, 18 * NT);
  bn_prelu_kernel<<<27000, 256, 0, stream>>>(ERB2, S_CE2, ce2_g, ce2_bb, a_conv,
      12, 18 * NT, 6912000L, 1.f / 576000.f);

  local_conv_kernel<<<11250, 256, 0, stream>>>(ERB2, ce3_W, ce3_b, ERB3,
      nullptr, nullptr, nullptr, 0.f,
      12, 18, 15, 6, 6, 1, 3, 3, 0);
  stats_kernel<<<240, 256, 0, stream>>>(ERB3, S_CE3, 15, 6 * NT);
  bn_prelu_kernel<<<11250, 256, 0, stream>>>(ERB3, S_CE3, ce3_g, ce3_bb, a_conv,
      15, 6 * NT, 2880000L, 1.f / 192000.f);

  // 3) com branch (input BN fused into cc1 loads -> no com_e0 materialization)
  local_conv_kernel<<<40000, 256, 0, stream>>>(com_feat, cc1_W, cc1_b, C1,
      S_COM, com_g, com_b, 1.f / (16.f * 320.f * 2000.f),
      2, 320, 4, 80, 4, 2, 4, 2, 1);
  stats_kernel<<<64, 256, 0, stream>>>(C1, S_CC1, 4, 80 * NT);
  bn_prelu_kernel<<<40000, 256, 0, stream>>>(C1, S_CC1, cc1_g, cc1_bb, a_conv,
      4, 80 * NT, 10240000L, 1.f / 2560000.f);

  local_conv_kernel<<<40000, 256, 0, stream>>>(C1, cc2_W, cc2_b, C2,
      nullptr, nullptr, nullptr, 0.f,
      4, 80, 8, 40, 4, 1, 2, 2, 0);
  stats_kernel<<<128, 256, 0, stream>>>(C2, S_CC2, 8, 40 * NT);
  bn_prelu_kernel<<<40000, 256, 0, stream>>>(C2, S_CC2, cc2_g, cc2_bb, a_conv,
      8, 40 * NT, 10240000L, 1.f / 1280000.f);

  local_conv_kernel<<<11250, 256, 0, stream>>>(C2, cc3_W, cc3_b, C3,
      nullptr, nullptr, nullptr, 0.f,
      8, 40, 9, 10, 4, 1, 4, 2, 0);
  stats_kernel<<<144, 256, 0, stream>>>(C3, S_CC3, 9, 10 * NT);
  bn_prelu_kernel<<<11250, 256, 0, stream>>>(C3, S_CC3, cc3_g, cc3_bb, a_conv,
      9, 10 * NT, 2880000L, 1.f / 320000.f);

  // 4) grouped lin_in (+shuffle +PReLU) -> x_in
  lin_in_kernel<<<6000, 256, 0, stream>>>(ERB3, C3, lin_in_W, lin_in_b, a_in, XIN);

  // 5) gi = x_in @ Wih^T + bih  (WMMA, 2000 x 9 tiles)
  gi_wmma_kernel<<<dim3(2000, 9), 32, 0, stream>>>(XIN, gru_Wih, gru_bih, GI);

  // 6) GRU scan (single workgroup, WMMA recurrence + async gi staging)
  gru_kernel<<<1, 384, 0, stream>>>(GI, gru_Whh, gru_bhh, YS);

  // 7) skip + grouped lin_out (+shuffle +PReLU) -> cat_gru
  lin_out_kernel<<<10000, 256, 0, stream>>>(YS, XIN, lin_out_W, lin_out_b, a_out, CAT);
}
